// Interaction_65060164599845
// MI455X (gfx1250) — matrix operand order repaired and checked by reading, not compiled, not run on hardware
//
#include <hip/hip_runtime.h>
#include <hip/hip_bf16.h>
#include <math.h>

// ---------------------------------------------------------------------------
// TensorNet interaction layer for MI455X (gfx1250, wave32, WMMA + TDM).
//   K0 prep     : one-time weight convert f32->f16 + fragment-major swizzle
//   K1 decomp   : normalize X, decompose to 10 components (f16)
//   K2 chanlin  : 10 batched [M=10000,K=128]x[128,128] WMMA GEMMs (wt[0..2])
//                 weights staged to LDS via tensor_load_to_lds (TDM)
//   K3 edge     : fused 3-layer MLP (WMMA f16, 288KB LDS, TDM weight DMA)
//                 + cosine cutoff + structured I/A/S scatter (f32 atomics)
//   K4 interact : per-(atom,ch) 3x3 algebra -> new components (f16)
//   K5 chanlin  : same GEMM kernel with wt[3..5]
//   K6 final    : out = Xn + dX + dX@dX
// ---------------------------------------------------------------------------

#define NATOMS 10000
#define NEDGES 160000
#define HID    128
#define NRBF   64
#define NH     (NATOMS * HID)   // 1,280,000 (atom,channel) pairs

// pre-swizzled weight arena (halves) inside workspace
#define OFF_W1 0        //  128x64   =  8192 halves
#define OFF_W2 8192     //  256x128  = 32768 halves
#define OFF_W3 40960    //  2 chunks of 192x256 = 98304 halves
#define OFF_WT 139264   //  6 x 128x128 = 98304 halves
#define WSW_BYTES_OFF 128000000

typedef __attribute__((ext_vector_type(16))) _Float16 v16h;
typedef __attribute__((ext_vector_type(8)))  float    v8f;
typedef __attribute__((ext_vector_type(4)))  unsigned int u32x4;
typedef __attribute__((ext_vector_type(8)))  int          i32x8;
typedef __attribute__((ext_vector_type(4)))  int          i32x4;

static __device__ __forceinline__ v8f vzero() {
  v8f z = {0.f, 0.f, 0.f, 0.f, 0.f, 0.f, 0.f, 0.f};
  return z;
}

static __device__ __forceinline__ v8f wmma_f16(v16h a, v16h b, v8f c) {
  // D = A(16x32 f16) * B(32x16 f16) + C(16x16 f32)
  return __builtin_amdgcn_wmma_f32_16x16x32_f16(false, a, false, b,
                                                (short)0, c, false, false);
}

static __device__ __forceinline__ float siluf(float x) {
  return x * (1.f / (1.f + __expf(-x)));
}

// ---------------------------------------------------------------------------
// TDM: contiguous copy of nbytes (multiple of 8) from global f16 arena into
// LDS at byte offset ldsOff.  1-D tile, data_size = 8B units.  Issued by one
// wave (EXEC ignored, per-wave op); caller pairs with s_wait_tensorcnt.
// ---------------------------------------------------------------------------
static __device__ __forceinline__ void tdm_copy_to_lds(const _Float16* gsrc,
                                                       unsigned ldsOff,
                                                       unsigned nbytes) {
  unsigned long long ga = (unsigned long long)(size_t)gsrc;
  unsigned units = nbytes >> 3;  // 8-byte elements (<= 65535)
  u32x4 g0 = {
      1u,                                            // count=1 (valid, user)
      ldsOff,                                        // lds_addr (bytes)
      (unsigned)(ga & 0xffffffffu),                  // global_addr[31:0]
      (unsigned)((ga >> 32) & 0x1ffffffu) | (2u << 30)  // ga[56:32] | type=2
  };
  i32x8 g1 = {
      (int)(3u << 16),                 // wg_mask=0, data_size=3 (8B)
      (int)((units & 0xffffu) << 16),  // tensor_dim0[15:0] @ bits 63:48
      (int)((units >> 16) | (1u << 16)),  // tensor_dim0[31:16], tensor_dim1=1
      (int)((units & 0xffffu) << 16),  // tile_dim0 @ bits 127:112
      0,                               // tile_dim1=0, tile_dim2=0 (unused)
      (int)units,                      // tensor_dim0_stride[31:0]
      0, 0
  };
  i32x4 z4 = {0, 0, 0, 0};
#if __clang_major__ >= 23
  i32x8 z8 = {0, 0, 0, 0, 0, 0, 0, 0};
  __builtin_amdgcn_tensor_load_to_lds(g0, g1, z4, z4, z8, 0);
#else
  __builtin_amdgcn_tensor_load_to_lds(g0, g1, z4, z4, 0);
#endif
}

// ---------------------------------------------------------------------------
// Fragment layouts (CDNA5 ISA 16-bit A / B VGPR layouts).  Each (tile,kstep)
// block = 512 halves = 32 lanes x 16 halves -> one contiguous 32B read/lane.
// ---------------------------------------------------------------------------

// B(k_local, n_local): lane = n_local + 16*((k_local%16)/8),
//                      half j = 8*(k_local/16) + (k_local%8).
// Swizzle row-major W[outN][K] f32 -> fragment-major f16 (global dst).
static __device__ __forceinline__ void swizB_global(const float* __restrict__ W,
                                                    _Float16* __restrict__ dst,
                                                    int outN, int K) {
  const int NT = outN >> 4;
  const int total = outN * K;
  for (int idx = threadIdx.x; idx < total; idx += 256) {
    int b = idx >> 9;
    int r = idx & 511;
    int lane = r >> 4, j = r & 15;
    int ntile = b % NT, kstep = b / NT;
    int n = (ntile << 4) + (lane & 15);
    int k = (kstep << 5) + ((j >> 3) << 4) + ((lane >> 4) << 3) + (j & 7);
    dst[idx] = (_Float16)W[n * K + k];
  }
}

// A(m_local, k_local): lane = m_local + 16*((k_local%16)/8),
//                      half j = 8*(k_local/16) + (k_local%8).
static __device__ __forceinline__ void stageA_f32(const float* __restrict__ src,
                                                  int row0, int stride, int K,
                                                  int validRows, _Float16* dst) {
  const int total = 128 * K;
  for (int idx = threadIdx.x; idx < total; idx += 256) {
    int b = idx >> 9, r = idx & 511;
    int lane = r >> 4, j = r & 15;
    int mtile = b & 7, kstep = b >> 3;
    int m = (mtile << 4) + (lane & 15);
    int kl = ((j >> 3) << 4) + ((lane >> 4) << 3) + (j & 7);
    int k = (kstep << 5) + kl;
    float v = (m < validRows) ? src[(row0 + m) * stride + k] : 0.f;
    dst[idx] = (_Float16)v;
  }
}

static __device__ __forceinline__ void stageA_f16(const _Float16* __restrict__ src,
                                                  int row0, int stride, int K,
                                                  int validRows, _Float16* dst) {
  const int total = 128 * K;
  for (int idx = threadIdx.x; idx < total; idx += 256) {
    int b = idx >> 9, r = idx & 511;
    int lane = r >> 4, j = r & 15;
    int mtile = b & 7, kstep = b >> 3;
    int m = (mtile << 4) + (lane & 15);
    int kl = ((j >> 3) << 4) + ((lane >> 4) << 3) + (j & 7);
    int k = (kstep << 5) + kl;
    dst[idx] = (m < validRows) ? src[(row0 + m) * stride + k] : (_Float16)0.f;
  }
}

static __device__ __forceinline__ v16h ldA(const _Float16* base, int mtile,
                                           int kstep, int lane) {
  return *(const v16h*)(base + (((kstep << 3) + mtile) << 9) + (lane << 4));
}
static __device__ __forceinline__ v16h ldB(const _Float16* base, int NT,
                                           int ntile, int kstep, int lane) {
  return *(const v16h*)(base + ((kstep * NT + ntile) << 9) + (lane << 4));
}

// bias + silu on C tile, store into A-swizzled f16 LDS for the next layer.
static __device__ __forceinline__ void storeAct(_Float16* dstA, int mtile,
                                                v8f acc, int n0, int lane,
                                                const float* __restrict__ bias) {
  const int n = n0 + (lane & 15);
  const float bv = bias[n];
  const int kstep = n >> 5, kl = n & 31;
  const int dlaneHalf = ((kl & 15) >> 3) << 4;
  const int j = ((kl >> 4) << 3) + (kl & 7);
  const int blockBase = (((kstep << 3) + mtile) << 9) + j;
#pragma unroll
  for (int r = 0; r < 8; ++r) {
    int m = r + ((lane >> 4) << 3);
    float s = siluf(acc[r] + bv);
    dstA[blockBase + (((m & 15) + dlaneHalf) << 4)] = (_Float16)s;
  }
}

// recompose 3x3 from 10 components (c0=I scalar, c1..3 antisym, c4..9 sym)
static __device__ __forceinline__ void recomp3(float* M, const float* g) {
  M[0] = g[0] + g[4]; M[1] =  g[1] + g[5]; M[2] =  g[2] + g[6];
  M[3] = g[5] - g[1]; M[4] =  g[0] + g[7]; M[5] =  g[3] + g[8];
  M[6] = g[6] - g[2]; M[7] =  g[8] - g[3]; M[8] =  g[0] + g[9];
}

static __device__ __forceinline__ void mm3(const float* A, const float* B, float* C) {
#pragma unroll
  for (int i = 0; i < 3; ++i)
#pragma unroll
    for (int j = 0; j < 3; ++j)
      C[i * 3 + j] = A[i * 3 + 0] * B[0 + j] + A[i * 3 + 1] * B[3 + j] +
                     A[i * 3 + 2] * B[6 + j];
}

// ---------------------------------------------------------------------------
// K0: convert + swizzle all weights once into the f16 fragment-major arena
// ---------------------------------------------------------------------------
__global__ void k_prep(const float* __restrict__ w1, const float* __restrict__ w2,
                       const float* __restrict__ w3, const float* __restrict__ wt,
                       _Float16* __restrict__ wsw) {
  switch (blockIdx.x) {
    case 0: swizB_global(w1, wsw + OFF_W1, 128, 64); break;
    case 1: swizB_global(w2, wsw + OFF_W2, 256, 128); break;
    case 2: swizB_global(w3, wsw + OFF_W3, 192, 256); break;
    case 3: swizB_global(w3 + 192 * 256, wsw + OFF_W3 + 49152, 192, 256); break;
    default: {
      int p = blockIdx.x - 4;  // wt planes 0..5
      swizB_global(wt + p * HID * HID, wsw + OFF_WT + p * 16384, 128, 128);
    } break;
  }
}

// ---------------------------------------------------------------------------
// K1: normalize + decompose X into 10 f16 component planes
// ---------------------------------------------------------------------------
__global__ void k_decomp(const float* __restrict__ X, _Float16* __restrict__ xc) {
  int id = blockIdx.x * 256 + threadIdx.x;
  if (id >= NH) return;
  const float* t = X + (size_t)id * 9;
  float m[9];
  float nrm = 0.f;
#pragma unroll
  for (int k = 0; k < 9; ++k) { m[k] = t[k]; nrm += m[k] * m[k]; }
  float inv = 1.f / (nrm + 1.f);
#pragma unroll
  for (int k = 0; k < 9; ++k) m[k] *= inv;
  float c0 = (m[0] + m[4] + m[8]) * (1.f / 3.f);
  xc[0 * NH + id] = (_Float16)c0;
  xc[1 * NH + id] = (_Float16)(0.5f * (m[1] - m[3]));
  xc[2 * NH + id] = (_Float16)(0.5f * (m[2] - m[6]));
  xc[3 * NH + id] = (_Float16)(0.5f * (m[5] - m[7]));
  xc[4 * NH + id] = (_Float16)(m[0] - c0);
  xc[5 * NH + id] = (_Float16)(0.5f * (m[1] + m[3]));
  xc[6 * NH + id] = (_Float16)(0.5f * (m[2] + m[6]));
  xc[7 * NH + id] = (_Float16)(m[4] - c0);
  xc[8 * NH + id] = (_Float16)(0.5f * (m[5] + m[7]));
  xc[9 * NH + id] = (_Float16)(m[8] - c0);
}

// ---------------------------------------------------------------------------
// K2/K5: batched chan_lin.  grid = (ceil(N/128), 10 components).
// Weight tile DMA'd to LDS by TDM; activations staged/swizzled by all waves.
// ---------------------------------------------------------------------------
__global__ void __launch_bounds__(256)
k_chanlin(const _Float16* __restrict__ xc, const _Float16* __restrict__ wsw,
          float* __restrict__ yc, int wbase) {
  extern __shared__ char smem[];
  _Float16* wf = (_Float16*)smem;          // 16384 halves (32KB) @ byte 0
  _Float16* aA = (_Float16*)smem + 16384;  // 16384 halves (32KB) @ byte 32768
  const int c = blockIdx.y;
  const int wsel = wbase + ((c == 0) ? 0 : ((c < 4) ? 1 : 2));

  if (threadIdx.x < 32)  // wave 0 issues the weight DMA (per-wave op)
    tdm_copy_to_lds(wsw + OFF_WT + wsel * 16384, 0u, 32768u);

  const int row0 = blockIdx.x * 128;
  int valid = NATOMS - row0; if (valid > 128) valid = 128;
  stageA_f16(xc + (size_t)c * NH, row0, HID, HID, valid, aA);

  if (threadIdx.x < 32) __builtin_amdgcn_s_wait_tensorcnt(0);
  __syncthreads();

  const int w = threadIdx.x >> 5, lane = threadIdx.x & 31;
  v8f acc[8];
#pragma unroll
  for (int nt = 0; nt < 8; ++nt) acc[nt] = vzero();
#pragma unroll
  for (int ks = 0; ks < 4; ++ks) {
    v16h a = ldA(aA, w, ks, lane);
#pragma unroll
    for (int nt = 0; nt < 8; ++nt)
      acc[nt] = wmma_f16(a, ldB(wf, 8, nt, ks, lane), acc[nt]);
  }
  float* out = yc + (size_t)c * NH;
  const int mrow = row0 + (w << 4) + ((lane >> 4) << 3);
  const int ncol = lane & 15;
#pragma unroll
  for (int nt = 0; nt < 8; ++nt) {
#pragma unroll
    for (int r = 0; r < 8; ++r) {
      int atom = mrow + r;
      if (atom < NATOMS) out[atom * HID + (nt << 4) + ncol] = acc[nt][r];
    }
  }
}

// ---------------------------------------------------------------------------
// K3: fused edge MLP (3 WMMA layers, 288KB LDS, TDM weight DMA) + scatter.
// One workgroup per 128 edges; each wave owns 16 edge-rows through all layers.
// ---------------------------------------------------------------------------
__global__ void __launch_bounds__(256, 1)
k_edge(const float* __restrict__ attr, const float* __restrict__ ew,
       const int* __restrict__ ei, const _Float16* __restrict__ wsw,
       const float* __restrict__ b1, const float* __restrict__ b2,
       const float* __restrict__ b3,
       const float* __restrict__ yc, float* __restrict__ msg) {
  extern __shared__ char smem[];
  _Float16* S     = (_Float16*)smem;
  _Float16* h2A   = S;             // 32768 halves (64KB) @ byte 0
  _Float16* h1A   = S + 32768;     // 16384 halves (32KB) @ byte 65536
  _Float16* attrA = S + 49152;     //  8192 halves (16KB) @ byte 98304
  _Float16* w1f   = S + 57344;     //  8192 halves (16KB) @ byte 114688
  _Float16* w2f   = S + 65536;     // 32768 halves (64KB) @ byte 131072
  _Float16* w3f   = S + 49152;     // 49152 halves (96KB) @ byte 98304 (alias)
  _Float16* eaL   = S + 98304;     // 49152 halves (96KB) @ byte 196608
  float*    cutL  = (float*)(smem + 294912);  // 128 floats

  const int e0 = blockIdx.x * 128;
  const int tid = threadIdx.x;

  if (tid < 32) {  // wave 0: DMA layer-1/2 weights while others stage attr
    tdm_copy_to_lds(wsw + OFF_W1, 114688u, 16384u);
    tdm_copy_to_lds(wsw + OFF_W2, 131072u, 65536u);
  }
  if (tid < 128) {
    float r = ew[e0 + tid];
    float cc = 0.5f * (__cosf(r * 0.62831853f) + 1.f);  // pi/5
    cutL[tid] = (r < 5.0f) ? cc : 0.f;
  }
  stageA_f32(attr, e0, NRBF, NRBF, 128, attrA);
  if (tid < 32) __builtin_amdgcn_s_wait_tensorcnt(0);
  __syncthreads();

  const int w = tid >> 5, lane = tid & 31;

  // ---- layer 1: [128x64] @ w1^T -> silu -> h1A ----
  {
    v8f acc[8];
#pragma unroll
    for (int nt = 0; nt < 8; ++nt) acc[nt] = vzero();
#pragma unroll
    for (int ks = 0; ks < 2; ++ks) {
      v16h a = ldA(attrA, w, ks, lane);
#pragma unroll
      for (int nt = 0; nt < 8; ++nt)
        acc[nt] = wmma_f16(a, ldB(w1f, 8, nt, ks, lane), acc[nt]);
    }
#pragma unroll
    for (int nt = 0; nt < 8; ++nt) storeAct(h1A, w, acc[nt], nt << 4, lane, b1);
  }

  // ---- layer 2: [128x128] @ w2^T -> silu -> h2A (two 128-col halves) ----
  // Wave w reads only rows it produced itself: no cross-wave barrier needed.
#pragma unroll
  for (int half = 0; half < 2; ++half) {
    v8f acc[8];
#pragma unroll
    for (int nt = 0; nt < 8; ++nt) acc[nt] = vzero();
#pragma unroll
    for (int ks = 0; ks < 4; ++ks) {
      v16h a = ldA(h1A, w, ks, lane);
#pragma unroll
      for (int nt = 0; nt < 8; ++nt)
        acc[nt] = wmma_f16(a, ldB(w2f, 16, (half << 3) + nt, ks, lane), acc[nt]);
    }
#pragma unroll
    for (int nt = 0; nt < 8; ++nt)
      storeAct(h2A, w, acc[nt], ((half << 3) + nt) << 4, lane, b2);
  }
  __syncthreads();  // w2f/attrA region about to be overwritten by w3 chunks

  // ---- layer 3: [128x256] @ w3^T in two 192-col chunks -> ea (LDS) ----
  for (int ch = 0; ch < 2; ++ch) {
    if (tid < 32) {
      tdm_copy_to_lds(wsw + OFF_W3 + ch * 49152, 98304u, 98304u);
      __builtin_amdgcn_s_wait_tensorcnt(0);
    }
    __syncthreads();
    v8f acc[12];
#pragma unroll
    for (int nt = 0; nt < 12; ++nt) acc[nt] = vzero();
#pragma unroll
    for (int ks = 0; ks < 8; ++ks) {
      v16h a = ldA(h2A, w, ks, lane);
#pragma unroll
      for (int nt = 0; nt < 12; ++nt)
        acc[nt] = wmma_f16(a, ldB(w3f, 12, nt, ks, lane), acc[nt]);
    }
    const int mloc = (w << 4) + ((lane >> 4) << 3);
    const int nbase = ch * 192 + (lane & 15);
#pragma unroll
    for (int nt = 0; nt < 12; ++nt) {
      int n = nbase + (nt << 4);
      float bv = b3[n];
#pragma unroll
      for (int r = 0; r < 8; ++r) {
        int eL = mloc + r;
        float s = siluf(acc[nt][r] + bv) * cutL[eL];
        eaL[eL * 384 + n] = (_Float16)s;
      }
    }
    __syncthreads();  // ea visible; safe to overwrite w3f next iteration
  }

  // ---- scatter: msg[dst] += ea(c) * comp[src] over 10 components ----
  for (int p = tid; p < 128 * HID; p += 256) {
    const int eL = p & 127, h = p >> 7;
    const int e = e0 + eL;
    const int dst = ei[e];           // edge_index[0]
    const int src = ei[NEDGES + e];  // edge_index[1]
    const float f0 = (float)eaL[eL * 384 + h * 3 + 0];
    const float f1 = (float)eaL[eL * 384 + h * 3 + 1];
    const float f2 = (float)eaL[eL * 384 + h * 3 + 2];
    const int si = src * HID + h, di = dst * HID + h;
    unsafeAtomicAdd(msg + di, f0 * yc[si]);
#pragma unroll
    for (int c = 1; c < 4; ++c)
      unsafeAtomicAdd(msg + c * NH + di, f1 * yc[c * NH + si]);
#pragma unroll
    for (int c = 4; c < 10; ++c)
      unsafeAtomicAdd(msg + c * NH + di, f2 * yc[c * NH + si]);
  }
}

// ---------------------------------------------------------------------------
// K4: T = M@Y + Y@M, decompose, scale by 1/(|T|^2+1), emit f16 components
// ---------------------------------------------------------------------------
__global__ void k_interact(const float* __restrict__ msg,
                           const float* __restrict__ yc,
                           _Float16* __restrict__ xc) {
  int id = blockIdx.x * 256 + threadIdx.x;
  if (id >= NH) return;
  float g[10], y[10];
#pragma unroll
  for (int c = 0; c < 10; ++c) { g[c] = msg[c * NH + id]; y[c] = yc[c * NH + id]; }
  float M[9], Y[9], A[9], B[9], T[9];
  recomp3(M, g);
  recomp3(Y, y);
  mm3(M, Y, A);
  mm3(Y, M, B);
  float tn = 0.f;
#pragma unroll
  for (int k = 0; k < 9; ++k) { T[k] = A[k] + B[k]; tn += T[k] * T[k]; }
  float inv = 1.f / (tn + 1.f);
  float c0 = (T[0] + T[4] + T[8]) * (1.f / 3.f);
  xc[0 * NH + id] = (_Float16)(c0 * inv);
  xc[1 * NH + id] = (_Float16)(0.5f * (T[1] - T[3]) * inv);
  xc[2 * NH + id] = (_Float16)(0.5f * (T[2] - T[6]) * inv);
  xc[3 * NH + id] = (_Float16)(0.5f * (T[5] - T[7]) * inv);
  xc[4 * NH + id] = (_Float16)((T[0] - c0) * inv);
  xc[5 * NH + id] = (_Float16)(0.5f * (T[1] + T[3]) * inv);
  xc[6 * NH + id] = (_Float16)(0.5f * (T[2] + T[6]) * inv);
  xc[7 * NH + id] = (_Float16)((T[4] - c0) * inv);
  xc[8 * NH + id] = (_Float16)(0.5f * (T[5] + T[7]) * inv);
  xc[9 * NH + id] = (_Float16)((T[8] - c0) * inv);
}

// ---------------------------------------------------------------------------
// K6: out = Xn + dX + dX@dX  (Xn recomputed from original X)
// ---------------------------------------------------------------------------
__global__ void k_final(const float* __restrict__ dxc,
                        const float* __restrict__ X, float* __restrict__ out) {
  int id = blockIdx.x * 256 + threadIdx.x;
  if (id >= NH) return;
  float d[10];
#pragma unroll
  for (int c = 0; c < 10; ++c) d[c] = dxc[c * NH + id];
  float D[9], P[9];
  recomp3(D, d);
  mm3(D, D, P);
  const float* t = X + (size_t)id * 9;
  float nrm = 0.f;
#pragma unroll
  for (int k = 0; k < 9; ++k) nrm += t[k] * t[k];
  float inv = 1.f / (nrm + 1.f);
  float* o = out + (size_t)id * 9;
#pragma unroll
  for (int k = 0; k < 9; ++k) o[k] = t[k] * inv + D[k] + P[k];
}

// ---------------------------------------------------------------------------
extern "C" void kernel_launch(void* const* d_in, const int* in_sizes, int n_in,
                              void* d_out, int out_size, void* d_ws, size_t ws_size,
                              hipStream_t stream) {
  (void)in_sizes; (void)n_in; (void)out_size; (void)ws_size;
  const float* X    = (const float*)d_in[0];
  const int*   ei   = (const int*)d_in[1];   // int32 per harness convention
  const float* ew   = (const float*)d_in[2];
  const float* attr = (const float*)d_in[3];
  const float* w1   = (const float*)d_in[4];
  const float* b1   = (const float*)d_in[5];
  const float* w2   = (const float*)d_in[6];
  const float* b2   = (const float*)d_in[7];
  const float* w3   = (const float*)d_in[8];
  const float* b3   = (const float*)d_in[9];
  const float* wt   = (const float*)d_in[10];

  char* ws = (char*)d_ws;
  _Float16* xc  = (_Float16*)ws;                   // 10*NH f16 = 25.6 MB
  float*    yc  = (float*)(ws + 25600000);         // 10*NH f32 = 51.2 MB
  float*    msg = (float*)(ws + 76800000);         // 10*NH f32 = 51.2 MB
  _Float16* wsw = (_Float16*)(ws + WSW_BYTES_OFF); // swizzled weights, 475 KB

  float* out = (float*)d_out;

  const int pgBlocks = (NH + 255) / 256;        // 5000
  const dim3 clGrid((NATOMS + 127) / 128, 10);  // 79 x 10

  k_prep<<<10, 256, 0, stream>>>(w1, w2, w3, wt, wsw);
  k_decomp<<<pgBlocks, 256, 0, stream>>>(X, xc);
  k_chanlin<<<clGrid, 256, 65536, stream>>>(xc, wsw, yc, 0);
  hipMemsetAsync(msg, 0, (size_t)10 * NH * sizeof(float), stream);
  k_edge<<<NEDGES / 128, 256, 295424, stream>>>(attr, ew, ei, wsw,
                                                b1, b2, b3, yc, msg);
  k_interact<<<pgBlocks, 256, 0, stream>>>(msg, yc, xc);
  k_chanlin<<<clGrid, 256, 65536, stream>>>(xc, wsw, yc, 3);
  k_final<<<pgBlocks, 256, 0, stream>>>(yc, X, out);
}
